// CustomMamba_47949014893047
// MI455X (gfx1250) — matrix operand order repaired and checked by reading, not compiled, not run on hardware
//
#include <hip/hip_runtime.h>
#include <math.h>

// ---------------------------------------------------------------------------
// Mamba-style 2-layer scan for MI455X (gfx1250, wave32, WMMA).
// All projections are f16-operand / f32-accumulate WMMA GEMMs with a
// barrier-free, LDS-free inner loop: activations are produced in f16 by the
// upstream kernels, so each K-chunk is 4 contiguous vector loads + 1
// v_wmma_f32_16x16x32_f16. Recurrent state lives in-place in d_out.
// ---------------------------------------------------------------------------

typedef __attribute__((ext_vector_type(16))) _Float16 v16h;
typedef __attribute__((ext_vector_type(8)))  _Float16 v8h;
typedef __attribute__((ext_vector_type(8)))  float    v8f;
typedef __attribute__((ext_vector_type(4)))  float    v4f;
typedef __attribute__((ext_vector_type(4)))  _Float16 v4h;

#define TT   32
#define BSZ  256
#define DM   512
#define DI   1024
#define DST  16
#define DTR  32
#define NL   2
#define STT  (4 + 16)   // conv(4) + ssm(16) per channel

// ---------------------------------------------------------------------------
// GEMM:  C[M,N](f32) = act( A[M,K](f16) * W[N,K]^T(f16) + bias )
// One 16x16 tile per wave, 4 waves (64 N) per 128-thread block.
// Fragment layouts used here (contiguous k, same order in A and B):
//   A lane(half,r):  af[e]    = A[r, k0 + 16*half + e]
//   B lane(half,r):  bf[e]    = W[n, k0 + 16*half + e]
// ACT: 0=none, 1=softplus.  ACCUM: C += result.  WRITE_H: also emit f16 copy.
// ---------------------------------------------------------------------------
template<int ACT, int HAS_BIAS, int ACCUM, int WRITE_H>
__global__ void __launch_bounds__(128)
gemm_wmma_h(const _Float16* __restrict__ A, const _Float16* __restrict__ W,
            const float* __restrict__ bias, float* __restrict__ C,
            _Float16* __restrict__ Ch, int M, int N, int K, int lda)
{
  const int tid  = threadIdx.x;
  const int wave = tid >> 5;
  const int lane = tid & 31;
  const int hlf  = lane >> 4;
  const int r    = lane & 15;
  const int m0   = blockIdx.y * 16;
  const int n0   = blockIdx.x * 64 + wave * 16;

  const _Float16* arow = A + (size_t)(m0 + r) * lda + 16 * hlf;
  const _Float16* wrow = W + (size_t)(n0 + r) * K + 16 * hlf;

  v8f acc = {};
#pragma unroll 2
  for (int k0 = 0; k0 < K; k0 += 32) {
    v8h  alo = *(const v8h*)(arow + k0);
    v8h  ahi = *(const v8h*)(arow + k0 + 8);
    v16h bf  = *(const v16h*)(wrow + k0);
    v16h af;
#pragma unroll
    for (int e = 0; e < 8; ++e) { af[e] = alo[e]; af[e + 8] = ahi[e]; }
    acc = __builtin_amdgcn_wmma_f32_16x16x32_f16(
        /*neg_a=*/false, af, /*neg_b=*/false, bf,
        /*c_mod=*/(short)0, acc, /*reuse_a=*/false, /*reuse_b=*/false);
  }

  const int col = n0 + r;
  float bv = HAS_BIAS ? bias[col] : 0.0f;
#pragma unroll
  for (int v = 0; v < 8; ++v) {
    int row = m0 + v + 8 * hlf;
    float val = acc[v] + bv;
    if (ACT == 1) val = (val > 20.0f) ? val : log1pf(__expf(val));
    if (ACCUM) val += C[(size_t)row * N + col];
    C[(size_t)row * N + col] = val;
    if (WRITE_H) Ch[(size_t)row * N + col] = (_Float16)val;
  }
}

// ---------------------------------------------------------------------------
// LayerNorm over D=512; one wave32 per row; f32 math, f16 output (GEMM A).
// ---------------------------------------------------------------------------
__global__ void __launch_bounds__(256)
layernorm_rows(const float* __restrict__ x, const float* __restrict__ w,
               const float* __restrict__ b, _Float16* __restrict__ y)
{
  const int wave = threadIdx.x >> 5;
  const int lane = threadIdx.x & 31;
  const int row  = blockIdx.x * 8 + wave;
  const float* xr = x + (size_t)row * DM;
  float v[16];
  float s = 0.0f;
#pragma unroll
  for (int i = 0; i < 16; ++i) { v[i] = xr[lane + i * 32]; s += v[i]; }
#pragma unroll
  for (int m = 16; m >= 1; m >>= 1) s += __shfl_xor(s, m, 32);
  float mu = s * (1.0f / (float)DM);
  float q = 0.0f;
#pragma unroll
  for (int i = 0; i < 16; ++i) { float d = v[i] - mu; q += d * d; }
#pragma unroll
  for (int m = 16; m >= 1; m >>= 1) q += __shfl_xor(q, m, 32);
  float rstd = rsqrtf(q * (1.0f / (float)DM) + 1e-5f);
  _Float16* yr = y + (size_t)row * DM;
#pragma unroll
  for (int i = 0; i < 16; ++i) {
    int c = lane + i * 32;
    yr[c] = (_Float16)((v[i] - mu) * rstd * w[c] + b[c]);
  }
}

// ---------------------------------------------------------------------------
// Depthwise conv-4 state shift + SiLU.  states layout: [l][b][DI][20].
// Emits xconv in f32 (SSM) and f16 (x_proj GEMM A).
// ---------------------------------------------------------------------------
__global__ void __launch_bounds__(256)
conv_silu_step(const float* __restrict__ xz, float* __restrict__ states,
               const float* __restrict__ conv_w, const float* __restrict__ conv_b,
               float* __restrict__ xconv, _Float16* __restrict__ xconv_h, int l)
{
  int idx = blockIdx.x * 256 + threadIdx.x;     // b*DI + d
  if (idx >= BSZ * DI) return;
  int b = idx >> 10, d = idx & (DI - 1);
  float xc = xz[(size_t)b * (2 * DI) + d];
  float* sp = states + ((size_t)(l * BSZ + b) * DI + d) * STT;
  float c1 = sp[1], c2 = sp[2], c3 = sp[3];
  sp[0] = c1; sp[1] = c2; sp[2] = c3; sp[3] = xc;
  const float* cw = conv_w + (size_t)(l * DI + d) * 4;
  float a = c1 * cw[0] + c2 * cw[1] + c3 * cw[2] + xc * cw[3]
          + conv_b[l * DI + d];
  float sig = 1.0f / (1.0f + __expf(-a));
  float r = a * sig;
  xconv[idx]   = r;
  xconv_h[idx] = (_Float16)r;
}

// ---------------------------------------------------------------------------
// Selective-SSM update + output gate:  per (b, d) channel, 16 states.
//   ss = ss*exp(dt * -exp(A_log)) + (xconv*dt)*B ;  y = <ss,C> + D*xconv
//   yz = y * silu(z)   (f16 out: only consumed as out_proj GEMM A)
// ---------------------------------------------------------------------------
__global__ void __launch_bounds__(256)
ssm_step(const float* __restrict__ xdb, const float* __restrict__ dt,
         const float* __restrict__ xconv, const float* __restrict__ A_log,
         const float* __restrict__ D_skip, const float* __restrict__ xz,
         float* __restrict__ states, _Float16* __restrict__ yz, int l)
{
  int idx = blockIdx.x * 256 + threadIdx.x;     // b*DI + d
  if (idx >= BSZ * DI) return;
  int b = idx >> 10, d = idx & (DI - 1);
  float dtv = dt[idx];
  float xc  = xconv[idx];
  float xdt = xc * dtv;
  const float* al = A_log + (size_t)d * DST;          // pre-offset by layer
  const float* Bm = xdb + (size_t)b * (DTR + 2 * DST) + DTR;
  const float* Cm = Bm + DST;
  float* sp = states + ((size_t)(l * BSZ + b) * DI + d) * STT + 4;
  float y = 0.0f;
#pragma unroll
  for (int s = 0; s < DST; ++s) {
    float dA = __expf(-dtv * __expf(al[s]));
    float ss = sp[s] * dA + xdt * Bm[s];
    sp[s] = ss;
    y += ss * Cm[s];
  }
  y += D_skip[d] * xc;
  float z = xz[(size_t)b * (2 * DI) + DI + d];
  float sig = 1.0f / (1.0f + __expf(-z));
  yz[idx] = (_Float16)(y * (z * sig));
}

// ---------------------------------------------------------------------------
// Vectorized f32 -> f16 conversion (4 elements / thread).
// ---------------------------------------------------------------------------
__global__ void cvt_f32_f16_x4(const float* __restrict__ s,
                               _Float16* __restrict__ d, int n4)
{
  int i = blockIdx.x * 256 + threadIdx.x;
  if (i >= n4) return;
  v4f a = ((const v4f*)s)[i];
  v4h h;
#pragma unroll
  for (int e = 0; e < 4; ++e) h[e] = (_Float16)a[e];
  ((v4h*)d)[i] = h;
}

// ---------------------------------------------------------------------------
extern "C" void kernel_launch(void* const* d_in, const int* in_sizes, int n_in,
                              void* d_out, int out_size, void* d_ws, size_t ws_size,
                              hipStream_t stream)
{
  const float* x          = (const float*)d_in[0];
  const float* rnn_states = (const float*)d_in[1];
  const float* ip_w       = (const float*)d_in[2];
  const float* ip_b       = (const float*)d_in[3];
  const float* op_w       = (const float*)d_in[4];
  const float* op_b       = (const float*)d_in[5];
  const float* in_proj_w  = (const float*)d_in[6];
  const float* conv_w     = (const float*)d_in[7];
  const float* conv_b     = (const float*)d_in[8];
  const float* x_proj_w   = (const float*)d_in[9];
  const float* dt_proj_w  = (const float*)d_in[10];
  const float* dt_proj_b  = (const float*)d_in[11];
  const float* A_log      = (const float*)d_in[12];
  const float* D_skip     = (const float*)d_in[13];
  const float* out_proj_w = (const float*)d_in[14];
  const float* norm_w     = (const float*)d_in[15];
  const float* norm_b     = (const float*)d_in[16];
  const float* normf_w    = (const float*)d_in[17];
  const float* normf_b    = (const float*)d_in[18];

  float* out    = (float*)d_out;                        // [T,B,DM]
  float* states = out + (size_t)TT * BSZ * DM;          // [L,B,DI,20]

  // ---- workspace carve-out ------------------------------------------------
  char* p = (char*)d_ws;
  auto carve = [&](size_t bytes) -> void* {
    void* r = (void*)p;
    p += (bytes + 255) & ~(size_t)255;
    return r;
  };
  float*    resid   = (float*)   carve((size_t)BSZ * DM * 4);
  float*    xzb     = (float*)   carve((size_t)BSZ * 2 * DI * 4);
  float*    xconv   = (float*)   carve((size_t)BSZ * DI * 4);
  float*    xdb     = (float*)   carve((size_t)BSZ * (DTR + 2 * DST) * 4);
  float*    dtb     = (float*)   carve((size_t)BSZ * DI * 4);
  _Float16* x_h     = (_Float16*)carve((size_t)TT * BSZ * DM * 2);
  _Float16* hn_h    = (_Float16*)carve((size_t)BSZ * DM * 2);
  _Float16* xconv_h = (_Float16*)carve((size_t)BSZ * DI * 2);
  _Float16* xdb_h   = (_Float16*)carve((size_t)BSZ * (DTR + 2 * DST) * 2);
  _Float16* yz_h    = (_Float16*)carve((size_t)BSZ * DI * 2);
  _Float16* ys_h    = (_Float16*)carve((size_t)BSZ * DM * 2);
  _Float16* ip_w_h  = (_Float16*)carve((size_t)DM * DM * 2);
  _Float16* op_w_h  = (_Float16*)carve((size_t)DM * DM * 2);
  _Float16* inp_w_h = (_Float16*)carve((size_t)NL * 2 * DI * DM * 2);
  _Float16* xp_w_h  = (_Float16*)carve((size_t)NL * (DTR + 2 * DST) * DI * 2);
  _Float16* dtp_w_h = (_Float16*)carve((size_t)NL * DI * DTR * 2);
  _Float16* out_w_h = (_Float16*)carve((size_t)NL * DM * DI * 2);

  // ---- init recurrent state in-place in d_out -----------------------------
  hipMemcpyAsync(states, rnn_states, sizeof(float) * (size_t)NL * BSZ * DI * STT,
                 hipMemcpyDeviceToDevice, stream);

  // ---- one-time f32 -> f16 conversions (L2-resident afterwards) -----------
  auto cvt = [&](const float* s, _Float16* d, int n) {
    cvt_f32_f16_x4<<<dim3((n / 4 + 255) / 256), dim3(256), 0, stream>>>(s, d, n / 4);
  };
  cvt(x,          x_h,     TT * BSZ * DM);
  cvt(ip_w,       ip_w_h,  DM * DM);
  cvt(op_w,       op_w_h,  DM * DM);
  cvt(in_proj_w,  inp_w_h, NL * 2 * DI * DM);
  cvt(x_proj_w,   xp_w_h,  NL * (DTR + 2 * DST) * DI);
  cvt(dt_proj_w,  dtp_w_h, NL * DI * DTR);
  cvt(out_proj_w, out_w_h, NL * DM * DI);

  const dim3 blk128(128), blk256(256);
  const dim3 gLN(BSZ / 8);
  const dim3 gElem((BSZ * DI) / 256);

  for (int t = 0; t < TT; ++t) {
    // resid = x_t @ ip_w^T + ip_b        (M=256, N=512, K=512)
    gemm_wmma_h<0, 1, 0, 0><<<dim3(DM / 64, BSZ / 16), blk128, 0, stream>>>(
        x_h + (size_t)t * BSZ * DM, ip_w_h, ip_b, resid, nullptr,
        BSZ, DM, DM, DM);

    for (int l = 0; l < NL; ++l) {
      // hn = LN(resid)  (f16 out)
      layernorm_rows<<<gLN, blk256, 0, stream>>>(
          resid, norm_w + l * DM, norm_b + l * DM, hn_h);

      // xz = hn @ in_proj_w[l]^T         (M=256, N=2048, K=512)
      gemm_wmma_h<0, 0, 0, 0><<<dim3((2 * DI) / 64, BSZ / 16), blk128, 0, stream>>>(
          hn_h, inp_w_h + (size_t)l * 2 * DI * DM, nullptr, xzb, nullptr,
          BSZ, 2 * DI, DM, DM);

      // conv state shift + SiLU (f32 + f16 outputs)
      conv_silu_step<<<gElem, blk256, 0, stream>>>(
          xzb, states, conv_w, conv_b, xconv, xconv_h, l);

      // xdb = xconv @ x_proj_w[l]^T      (M=256, N=64, K=1024) + f16 copy
      gemm_wmma_h<0, 0, 0, 1><<<dim3((DTR + 2 * DST) / 64, BSZ / 16), blk128, 0, stream>>>(
          xconv_h, xp_w_h + (size_t)l * (DTR + 2 * DST) * DI, nullptr,
          xdb, xdb_h, BSZ, DTR + 2 * DST, DI, DI);

      // dt = softplus(xdb[:, :32] @ dt_proj_w[l]^T + dt_proj_b[l])
      gemm_wmma_h<1, 1, 0, 0><<<dim3(DI / 64, BSZ / 16), blk128, 0, stream>>>(
          xdb_h, dtp_w_h + (size_t)l * DI * DTR, dt_proj_b + l * DI,
          dtb, nullptr, BSZ, DI, DTR, DTR + 2 * DST);

      // SSM recurrence + gate (f16 out)
      ssm_step<<<gElem, blk256, 0, stream>>>(
          xdb, dtb, xconv, A_log + (size_t)l * DI * DST, D_skip + l * DI,
          xzb, states, yz_h, l);

      // resid += yz @ out_proj_w[l]^T    (M=256, N=512, K=1024)
      gemm_wmma_h<0, 0, 1, 0><<<dim3(DM / 64, BSZ / 16), blk128, 0, stream>>>(
          yz_h, out_w_h + (size_t)l * DM * DI, nullptr, resid, nullptr,
          BSZ, DM, DI, DI);
    }

    // ys_t = LN_f(resid);  out_t = ys_t @ op_w^T + op_b
    layernorm_rows<<<gLN, blk256, 0, stream>>>(resid, normf_w, normf_b, ys_h);
    gemm_wmma_h<0, 1, 0, 0><<<dim3(DM / 64, BSZ / 16), blk128, 0, stream>>>(
        ys_h, op_w_h, op_b, out + (size_t)t * BSZ * DM, nullptr,
        BSZ, DM, DM, DM);
  }
}
